// GraphEncoder_11141145166298
// MI455X (gfx1250) — matrix operand order repaired and checked by reading, not compiled, not run on hardware
//
#include <hip/hip_runtime.h>
#include <hip/hip_bf16.h>

// Workspace layout (≈7.5 MB needed):
//   nbr  : N*32 int      (kNN indices, padded 20->32 by duplicating nbr 0)
//   Wb0/1/2 : bf16 weights, rows x KFP, zero-padded K-dim
//   h0   : N*64  f32
//   h1   : N*128 f32

typedef __attribute__((ext_vector_type(16))) __bf16 v16bf;
typedef __attribute__((ext_vector_type(8)))  __bf16 v8bf;
typedef __attribute__((ext_vector_type(4)))  __bf16 v4bf;
typedef __attribute__((ext_vector_type(8)))  float  v8f;

#define KNN  20
#define NPAD 32

// ---------------------------------------------------------------- kNN -------
__global__ __launch_bounds__(256) void knn_kernel(const float* __restrict__ x,
                                                  int* __restrict__ nbr, int N) {
    const int i = blockIdx.x * blockDim.x + threadIdx.x;
    __shared__ float tx[256 * 3];
    const float xi0 = x[i * 3 + 0], xi1 = x[i * 3 + 1], xi2 = x[i * 3 + 2];
    float bd[KNN]; int bi[KNN];
#pragma unroll
    for (int k = 0; k < KNN; ++k) { bd[k] = __builtin_inff(); bi[k] = 0; }
    for (int base = 0; base < N; base += 256) {
        const int t = threadIdx.x;
        tx[t * 3 + 0] = x[(base + t) * 3 + 0];
        tx[t * 3 + 1] = x[(base + t) * 3 + 1];
        tx[t * 3 + 2] = x[(base + t) * 3 + 2];
        __syncthreads();
        for (int jj = 0; jj < 256; ++jj) {
            const int j = base + jj;
            const float d0 = xi0 - tx[jj * 3 + 0];
            const float d1 = xi1 - tx[jj * 3 + 1];
            const float d2 = xi2 - tx[jj * 3 + 2];
            const float d  = d0 * d0 + d1 * d1 + d2 * d2;
            if (j != i && d < bd[KNN - 1]) {              // insertion into sorted top-K
                int p = KNN - 1;
                while (p > 0 && bd[p - 1] > d) { bd[p] = bd[p - 1]; bi[p] = bi[p - 1]; --p; }
                bd[p] = d; bi[p] = j;
            }
        }
        __syncthreads();
    }
#pragma unroll
    for (int k = 0; k < KNN; ++k)  nbr[i * NPAD + k] = bi[k];
#pragma unroll
    for (int k = KNN; k < NPAD; ++k) nbr[i * NPAD + k] = bi[0];  // dup: max-safe pad
}

// ------------------------------------------- weight f32 -> bf16 (zero-pad K) -
__global__ void wpad_kernel(const float* __restrict__ W, unsigned short* __restrict__ Wb,
                            int rows, int cin2, int kfp) {
    const int t = blockIdx.x * blockDim.x + threadIdx.x;
    if (t >= rows * kfp) return;
    const int r = t / kfp, c = t % kfp;
    const float v = (c < cin2) ? W[r * cin2 + c] : 0.f;
    reinterpret_cast<__bf16*>(Wb)[t] = (__bf16)v;
}

// ---------------------------------------------------------------- EdgeConv --
// One wave per point. feat (NPAD x KFP bf16) staged in LDS; D = feat @ W^T via
// v_wmma_f32_16x16x32_bf16; epilogue: +bias, ReLU, max over 32 rows.
template <int CIN, int COUT, int KFP, int WAVES>
__global__ __launch_bounds__(WAVES * 32) void edgeconv_kernel(
    const float* __restrict__ xin, const int* __restrict__ nbr,
    const unsigned short* __restrict__ Wb_raw, const float* __restrict__ bias,
    float* __restrict__ out) {
    constexpr int LSTR = KFP + 8;  // +16B row pad: kill LDS bank conflicts
    extern __shared__ char smem_raw[];
    const __bf16* __restrict__ Wb = reinterpret_cast<const __bf16*>(Wb_raw);

    const int lane = threadIdx.x & 31;
    const int wv   = threadIdx.x >> 5;
    const int i    = blockIdx.x * WAVES + wv;
    __bf16* feat = reinterpret_cast<__bf16*>(smem_raw) + (size_t)wv * NPAD * LSTR;

    // ---- build feature rows: [x_center | x_nbr - x_center], zero-pad to KFP
    {
        const int m = lane;                          // each lane owns one row
        const int j = nbr[i * NPAD + m];
        const float* xc = xin + (size_t)i * CIN;
        const float* xj = xin + (size_t)j * CIN;
        __bf16* row = feat + m * LSTR;
        if constexpr ((CIN & 3) == 0) {
            // vector path: float4 loads, packed 4xbf16 (8B) LDS stores
#pragma unroll 4
            for (int c = 0; c < CIN; c += 4) {
                const float4 c4 = *(const float4*)(xc + c);
                const float4 j4 = *(const float4*)(xj + c);
                v4bf pc, pd;
                pc[0] = (__bf16)c4.x; pc[1] = (__bf16)c4.y;
                pc[2] = (__bf16)c4.z; pc[3] = (__bf16)c4.w;
                pd[0] = (__bf16)(j4.x - c4.x); pd[1] = (__bf16)(j4.y - c4.y);
                pd[2] = (__bf16)(j4.z - c4.z); pd[3] = (__bf16)(j4.w - c4.w);
                *(v4bf*)(row + c)       = pc;
                *(v4bf*)(row + CIN + c) = pd;
            }
        } else {
#pragma unroll
            for (int c = 0; c < CIN; ++c) {
                const float fc = xc[c];
                row[c]       = (__bf16)fc;
                row[CIN + c] = (__bf16)(xj[c] - fc);
            }
        }
#pragma unroll
        for (int c = 2 * CIN; c < KFP; ++c) row[c] = (__bf16)0.f;
    }
    __syncthreads();

    const int half   = lane >> 4;   // 0: lanes 0-15, 1: lanes 16-31
    const int nlocal = lane & 15;
    const int ak0    = half * 8;    // A k-offset inside a 32-wide K tile
    const int bk0    = half * 8;    // B k-offset inside a 32-wide K tile

    constexpr int KT = KFP / 32;
    constexpr int NT = COUT / 16;

    // ---- hoist ALL A fragments into registers once (reused for every n-tile)
    v16bf A0[KT], A1[KT];
#pragma unroll
    for (int kt = 0; kt < KT; ++kt) {
        const int kbase = kt * 32;
        const __bf16* a0p = feat + nlocal * LSTR + kbase + ak0;
        const __bf16* a1p = feat + (16 + nlocal) * LSTR + kbase + ak0;
        const v8bf a0lo = *(const v8bf*)(a0p);
        const v8bf a0hi = *(const v8bf*)(a0p + 16);
        const v8bf a1lo = *(const v8bf*)(a1p);
        const v8bf a1hi = *(const v8bf*)(a1p + 16);
        A0[kt] = __builtin_shufflevector(a0lo, a0hi,
            0, 1, 2, 3, 4, 5, 6, 7, 8, 9, 10, 11, 12, 13, 14, 15);
        A1[kt] = __builtin_shufflevector(a1lo, a1hi,
            0, 1, 2, 3, 4, 5, 6, 7, 8, 9, 10, 11, 12, 13, 14, 15);
    }

#pragma unroll 1
    for (int nt = 0; nt < NT; ++nt) {
        const int n = nt * 16 + nlocal;
        // prefetch next n-tile's weight row into cache (global_prefetch_b8)
        if (nt + 1 < NT)
            __builtin_prefetch(Wb + (size_t)(n + 16) * KFP, 0, 3);
        v8f acc0 = {};  // rows 0..15 (neighbors 0..15)
        v8f acc1 = {};  // rows 16..31
#pragma unroll
        for (int kt = 0; kt < KT; ++kt) {
            const int kbase = kt * 32;
            // B fragment: lane = column n, k = bk0..bk0+7 and bk0+16..bk0+23 of W row n (bf16)
            const __bf16* bp = Wb + (size_t)n * KFP + kbase + bk0;
            const v8bf blo = *(const v8bf*)(bp);
            const v8bf bhi = *(const v8bf*)(bp + 16);
            const v16bf B = __builtin_shufflevector(blo, bhi,
                0, 1, 2, 3, 4, 5, 6, 7, 8, 9, 10, 11, 12, 13, 14, 15);
            acc0 = __builtin_amdgcn_wmma_f32_16x16x32_bf16(false, A0[kt], false, B,
                                                           (short)0, acc0, false, false);
            acc1 = __builtin_amdgcn_wmma_f32_16x16x32_bf16(false, A1[kt], false, B,
                                                           (short)0, acc1, false, false);
        }
        // epilogue: bias + ReLU + max over the 32 neighbor rows.
        // C/D layout: lane owns column n, rows (half*8 + v); shfl_xor 16 merges halves.
        const float bn = bias[n];
        float cmax = 0.f;  // ReLU floor
#pragma unroll
        for (int v = 0; v < 8; ++v) {
            cmax = fmaxf(cmax, acc0[v] + bn);   // v_max_num_f32, dual-issue friendly
            cmax = fmaxf(cmax, acc1[v] + bn);
        }
        cmax = fmaxf(cmax, __shfl_xor(cmax, 16, 32));
        if (lane < 16) out[(size_t)i * COUT + n] = cmax;
    }
}

// ---------------------------------------------------------------- launch ----
extern "C" void kernel_launch(void* const* d_in, const int* in_sizes, int n_in,
                              void* d_out, int out_size, void* d_ws, size_t ws_size,
                              hipStream_t stream) {
    const float* x  = (const float*)d_in[0];
    const float* W0 = (const float*)d_in[1];
    const float* b0 = (const float*)d_in[2];
    const float* W1 = (const float*)d_in[3];
    const float* b1 = (const float*)d_in[4];
    const float* W2 = (const float*)d_in[5];
    const float* b2 = (const float*)d_in[6];
    float* out = (float*)d_out;
    const int N = in_sizes[0] / 3;  // 8192

    char* ws = (char*)d_ws;
    auto al = [](size_t o) { return (o + 255) & ~(size_t)255; };
    size_t off = 0;
    int* nbr = (int*)(ws + off);               off = al(off + (size_t)N * NPAD * sizeof(int));
    unsigned short* Wb0 = (unsigned short*)(ws + off); off = al(off + (size_t)64 * 32 * 2);
    unsigned short* Wb1 = (unsigned short*)(ws + off); off = al(off + (size_t)128 * 128 * 2);
    unsigned short* Wb2 = (unsigned short*)(ws + off); off = al(off + (size_t)256 * 256 * 2);
    float* h0 = (float*)(ws + off);            off = al(off + (size_t)N * 64 * sizeof(float));
    float* h1 = (float*)(ws + off);            off = al(off + (size_t)N * 128 * sizeof(float));

    knn_kernel<<<N / 256, 256, 0, stream>>>(x, nbr, N);

    wpad_kernel<<<(64 * 32 + 255) / 256, 256, 0, stream>>>(W0, Wb0, 64, 6, 32);
    wpad_kernel<<<(128 * 128 + 255) / 256, 256, 0, stream>>>(W1, Wb1, 128, 128, 128);
    wpad_kernel<<<(256 * 256 + 255) / 256, 256, 0, stream>>>(W2, Wb2, 256, 256, 256);

    // shared = WAVES * 32 rows * (KFP+8) bf16
    edgeconv_kernel<3, 64, 32, 4><<<N / 4, 128, 4 * 32 * (32 + 8) * 2, stream>>>(
        x, nbr, Wb0, b0, h0);
    edgeconv_kernel<64, 128, 128, 4><<<N / 4, 128, 4 * 32 * (128 + 8) * 2, stream>>>(
        h0, nbr, Wb1, b1, h1);
    edgeconv_kernel<128, 256, 256, 2><<<N / 2, 64, 2 * 32 * (256 + 8) * 2, stream>>>(
        h1, nbr, Wb2, b2, out);
}
